// HyperGCN_45896020525560
// MI455X (gfx1250) — matrix operand order repaired and checked by reading, not compiled, not run on hardware
//
#include <hip/hip_runtime.h>
#include <hip/hip_bf16.h>

typedef __attribute__((ext_vector_type(16))) _Float16 v16h;
typedef __attribute__((ext_vector_type(8)))  float    v8f;

// Problem constants (match reference)
#define BQ   64
#define TQ   64
#define NDQ  512
#define HQ   512
#define NQ   4096      // B*T
#define N3Q  12288     // 3*N
#define MQ   4288      // B*(T+3)
#define LHYP 3
#define KCONV 4

#define CDIV(a,b) (((a)+(b)-1)/(b))

// ---------------------------------------------------------------------------
// f32 GEMM via f16 WMMA (f32 accumulate): C = A[M,K] @ W[K,N] (+bias)(+R)
// One wave computes a 32x64 tile of C (2 M-frags x 4 N-frags, 8 WMMA/K-step).
// Requires M,K multiples of 32 and N multiple of 64 (true at all call sites).
// Templated on bias/residual presence -> branch-free vectorizable epilogue.
// ---------------------------------------------------------------------------
template <bool HAS_BIAS, bool HAS_R>
__global__ __launch_bounds__(32)
void k_gemm_wmma(const float* __restrict__ A, const float* __restrict__ W,
                 const float* __restrict__ bias, const float* __restrict__ R,
                 float* __restrict__ C, int Mr, int K, int Nc)
{
  const int lane = threadIdx.x;        // 0..31
  const int kh   = lane >> 4;          // half-wave id
  const int l16  = lane & 15;
  const int n0 = blockIdx.x * 64;
  const int m0 = blockIdx.y * 32;
  if (m0 >= Mr || n0 >= Nc) return;

  v8f acc[2][4];
#pragma unroll
  for (int i = 0; i < 2; ++i)
#pragma unroll
    for (int j = 0; j < 4; ++j) acc[i][j] = (v8f){};

  for (int kk = 0; kk < K; kk += 32) {
    // ---- A fragments: rows m0..m0+31, K slice [kk, kk+32) ----
    // lane layout: row = (lane&15); half e -> K = kk + kh*16 + e
    v16h afr[2];
    {
      const float* ar0 = A + (size_t)(m0 + l16)      * K + kk + kh * 16;
      const float* ar1 = A + (size_t)(m0 + 16 + l16) * K + kk + kh * 16;
      if (kk + 32 < K) {
        __builtin_prefetch(ar0 + 32, 0, 0);
        __builtin_prefetch(ar1 + 32, 0, 0);
      }
#pragma unroll
      for (int e = 0; e < 8; ++e) {
        afr[0][e]     = (_Float16)ar0[e];
        afr[0][e + 8] = (_Float16)ar0[e + 8];
        afr[1][e]     = (_Float16)ar1[e];
        afr[1][e + 8] = (_Float16)ar1[e + 8];
      }
    }
    // ---- B fragments: K rows [kk,kk+32), cols n0..n0+63 ----
    // lane layout: col = (lane&15) + 16*frag; half e -> K = kk + kh*16 + e
    v16h bfr[4];
    {
      const float* wr = W + (size_t)(kk + kh * 16) * Nc + n0 + l16;
#pragma unroll
      for (int e = 0; e < 16; ++e) {
        const float* wre = wr + (size_t)e * Nc;
        bfr[0][e] = (_Float16)wre[0];
        bfr[1][e] = (_Float16)wre[16];
        bfr[2][e] = (_Float16)wre[32];
        bfr[3][e] = (_Float16)wre[48];
      }
    }
#pragma unroll
    for (int i = 0; i < 2; ++i)
#pragma unroll
      for (int j = 0; j < 4; ++j)
        acc[i][j] = __builtin_amdgcn_wmma_f32_16x16x32_f16(
            false, afr[i], false, bfr[j], (short)0, acc[i][j], false, false);
  }

  // ---- store: C 16x16 f32 layout: row = kh*8 + r, col = lane&15 ----
#pragma unroll
  for (int i = 0; i < 2; ++i) {
#pragma unroll
    for (int r = 0; r < 8; ++r) {
      const int row = m0 + i * 16 + kh * 8 + r;
#pragma unroll
      for (int j = 0; j < 4; ++j) {
        const int col = n0 + j * 16 + l16;
        const size_t o = (size_t)row * Nc + col;
        float v = acc[i][j][r];
        if (HAS_BIAS) v += bias[col];
        if (HAS_R)    v += R[o];
        C[o] = v;
      }
    }
  }
}

// ---------------------------------------------------------------------------
// Stage 0: speaker id + l2 = l + spk_emb[spk]
// ---------------------------------------------------------------------------
__global__ __launch_bounds__(128)
void k_spk_l2(const float* __restrict__ l, const float* __restrict__ qmask,
              const float* __restrict__ spk_emb, float* __restrict__ l2,
              int* __restrict__ spk)
{
  const int n = blockIdx.x;               // 0..N-1, n = b*T + t
  const int b = n / TQ, t = n % TQ;
  const float q0 = qmask[((size_t)t * BQ + b) * 2 + 0];
  const float q1 = qmask[((size_t)t * BQ + b) * 2 + 1];
  const int s = (q1 > q0) ? 1 : 0;        // argmax, first-max tie-break
  if (threadIdx.x == 0) spk[n] = s;
  const float* se = spk_emb + (size_t)s * NDQ;
  const float* lr = l + (size_t)n * NDQ;
  float* o = l2 + (size_t)n * NDQ;
  for (int d = threadIdx.x; d < NDQ; d += 128) o[d] = lr[d] + se[d];
}

// feats [N3,ND] (modality-block layout) and ucat [N, 3*ND] (row concat)
__global__ __launch_bounds__(256)
void k_feats(const float* __restrict__ l2, const float* __restrict__ a,
             const float* __restrict__ v, float* __restrict__ feats,
             float* __restrict__ ucat)
{
  const size_t idx = (size_t)blockIdx.x * blockDim.x + threadIdx.x;
  if (idx >= (size_t)NQ * NDQ) return;
  const int n = (int)(idx / NDQ), d = (int)(idx % NDQ);
  const int b = n / TQ, t = n % TQ;
  const float lv = l2[idx], av = a[idx], vv = v[idx];
  const size_t base = (size_t)b * 3 * TQ * NDQ;
  feats[base + (size_t)(0 * TQ + t) * NDQ + d] = lv;
  feats[base + (size_t)(1 * TQ + t) * NDQ + d] = av;
  feats[base + (size_t)(2 * TQ + t) * NDQ + d] = vv;
  float* ur = ucat + (size_t)n * 3 * NDQ;
  ur[d] = lv; ur[NDQ + d] = av; ur[2 * NDQ + d] = vv;
}

__global__ __launch_bounds__(256)
void k_eattr(const float* __restrict__ attr1, const float* __restrict__ attr2,
             const float* __restrict__ type1, float* __restrict__ eattr)
{
  const size_t idx = (size_t)blockIdx.x * blockDim.x + threadIdx.x;
  if (idx >= (size_t)MQ * HQ) return;
  const int m = (int)(idx / HQ), h = (int)(idx % HQ);
  const float t1 = type1[m];
  eattr[idx] = attr1[h] * t1 + attr2[h] * (1.f - t1);
}

// ---------------------------------------------------------------------------
// Hypergraph conv pieces
// ---------------------------------------------------------------------------
__global__ __launch_bounds__(32)
void k_rowdot(const float* __restrict__ X, const float* __restrict__ vv,
              float* __restrict__ out, int ncols)
{
  const int row = blockIdx.x;
  const float* xr = X + (size_t)row * ncols;
  float s = 0.f;
  for (int c = threadIdx.x; c < ncols; c += 32) s += xr[c] * vv[c];
  for (int off = 16; off; off >>= 1) s += __shfl_xor(s, off, 32);
  if (threadIdx.x == 0) out[row] = s;
}

__global__ __launch_bounds__(256)
void k_degrees(const int* __restrict__ node, const int* __restrict__ edge,
               const float* __restrict__ w_e, float* __restrict__ Bdeg,
               float* __restrict__ Ddeg, int kinc)
{
  const int k = blockIdx.x * blockDim.x + threadIdx.x;
  if (k >= kinc) return;
  atomicAdd(&Bdeg[edge[k]], 1.f);
  atomicAdd(&Ddeg[node[k]], w_e[edge[k]]);
}

__global__ __launch_bounds__(256)
void k_incexp(const int* __restrict__ node, const int* __restrict__ edge,
              const float* __restrict__ sn, const float* __restrict__ se,
              float* __restrict__ eexp, float* __restrict__ z, int kinc)
{
  const int k = blockIdx.x * blockDim.x + threadIdx.x;
  if (k >= kinc) return;
  float x = sn[node[k]] + se[edge[k]];
  x = (x > 0.f) ? x : 0.2f * x;          // leaky_relu(0.2)
  const float e = expf(x);               // logits are O(0.1): safe w/o max-sub
  eexp[k] = e;
  atomicAdd(&z[node[k]], e);
}

// ef[edge] += (alpha*ew) * xp[node]    (8 h-values per thread)
__global__ __launch_bounds__(256)
void k_scatter_ef(const int* __restrict__ node, const int* __restrict__ edge,
                  const float* __restrict__ eexp, const float* __restrict__ z,
                  const float* __restrict__ ew, const float* __restrict__ xp,
                  float* __restrict__ ef, int kinc)
{
  const size_t idx = (size_t)blockIdx.x * blockDim.x + threadIdx.x;
  if (idx >= (size_t)kinc * (HQ / 8)) return;
  const int k = (int)(idx / (HQ / 8));
  const int hb = (int)(idx % (HQ / 8)) * 8;
  const float coef = eexp[k] / z[node[k]] * ew[k];
  const float* xr = xp + (size_t)node[k] * HQ + hb;
  float* er = ef + (size_t)edge[k] * HQ + hb;
#pragma unroll
  for (int j = 0; j < 8; ++j) atomicAdd(&er[j], coef * xr[j]);
}

__global__ __launch_bounds__(256)
void k_ef_scale(const float* __restrict__ w_e, const float* __restrict__ Bdeg,
                float* __restrict__ ef)
{
  const size_t idx = (size_t)blockIdx.x * blockDim.x + threadIdx.x;
  if (idx >= (size_t)MQ * HQ) return;
  const int m = (int)(idx / HQ);
  ef[idx] *= w_e[m] / fmaxf(Bdeg[m], 1e-12f);
}

// acc[node] += alpha * ef[edge]
__global__ __launch_bounds__(256)
void k_scatter_out(const int* __restrict__ node, const int* __restrict__ edge,
                   const float* __restrict__ eexp, const float* __restrict__ z,
                   const float* __restrict__ ef, float* __restrict__ acc, int kinc)
{
  const size_t idx = (size_t)blockIdx.x * blockDim.x + threadIdx.x;
  if (idx >= (size_t)kinc * (HQ / 8)) return;
  const int k = (int)(idx / (HQ / 8));
  const int hb = (int)(idx % (HQ / 8)) * 8;
  const float alpha = eexp[k] / z[node[k]];
  const float* er = ef + (size_t)edge[k] * HQ + hb;
  float* ar = acc + (size_t)node[k] * HQ + hb;
#pragma unroll
  for (int j = 0; j < 8; ++j) atomicAdd(&ar[j], alpha * er[j]);
}

__global__ __launch_bounds__(256)
void k_hyper_final(const float* __restrict__ acc, const float* __restrict__ Ddeg,
                   const float* __restrict__ bias, float* __restrict__ xout)
{
  const size_t idx = (size_t)blockIdx.x * blockDim.x + threadIdx.x;
  if (idx >= (size_t)N3Q * HQ) return;
  const int v = (int)(idx / HQ), h = (int)(idx % HQ);
  xout[idx] = acc[idx] / fmaxf(Ddeg[v], 1e-12f) + bias[h];
}

// ---------------------------------------------------------------------------
// Utterance graph pieces
// ---------------------------------------------------------------------------
__global__ __launch_bounds__(32)
void k_norminv(const float* __restrict__ u, float* __restrict__ ninv)
{
  const int n = blockIdx.x;
  const float* ur = u + (size_t)n * HQ;
  float s = 0.f;
  for (int c = threadIdx.x; c < HQ; c += 32) { const float x = ur[c]; s += x * x; }
  for (int off = 16; off; off >>= 1) s += __shfl_xor(s, off, 32);
  if (threadIdx.x == 0) ninv[n] = 1.f / fmaxf(sqrtf(s), 1e-12f);
}

__global__ __launch_bounds__(64)
void k_simargmax(const float* __restrict__ u, const float* __restrict__ ninv,
                 int* __restrict__ jn)
{
  const int n = blockIdx.x;               // utterance i of dialogue b
  const int b = n / TQ, i = n % TQ;
  __shared__ float ui[HQ];
  __shared__ float vals[TQ];
  const float* urow = u + (size_t)n * HQ;
  for (int d = threadIdx.x; d < HQ; d += 64) ui[d] = urow[d];
  __syncthreads();
  const int j = threadIdx.x;
  const float* uj = u + (size_t)(b * TQ + j) * HQ;
  float s = 0.f;
  for (int d = 0; d < HQ; ++d) s += ui[d] * uj[d];
  s *= ninv[n] * ninv[b * TQ + j];
  if (j == i) s = -1e9f;
  vals[j] = s;
  __syncthreads();
  if (threadIdx.x == 0) {
    float best = vals[0]; int bi = 0;
    for (int jj = 1; jj < TQ; ++jj) if (vals[jj] > best) { best = vals[jj]; bi = jj; }
    jn[n] = bi;                           // first-max tie-break like jnp.argmax
  }
}

// deg accumulation over edge lists. mode 0: w=1, mode 1: w = same-speaker
__global__ __launch_bounds__(256)
void k_deg_pairs(const int* __restrict__ src, const int* __restrict__ dst,
                 const int* __restrict__ spk, int mode, float* __restrict__ deg,
                 int nedges)
{
  const int k = blockIdx.x * blockDim.x + threadIdx.x;
  if (k >= nedges) return;
  if (mode == 1 && spk[src[k]] != spk[dst[k]]) return;
  atomicAdd(&deg[dst[k]], 1.f);
}

__global__ __launch_bounds__(256)
void k_deg_tk(const int* __restrict__ jn, float* __restrict__ deg)
{
  const int n = blockIdx.x * blockDim.x + threadIdx.x;
  if (n >= NQ) return;
  const int b = n / TQ;
  const int jg = b * TQ + jn[n];
  atomicAdd(&deg[jg], 1.f);
  atomicAdd(&deg[n], 1.f);
}

__global__ __launch_bounds__(256)
void k_agg_pairs(const int* __restrict__ src, const int* __restrict__ dst,
                 const int* __restrict__ spk, int mode,
                 const float* __restrict__ x, float* __restrict__ agg, int nedges)
{
  const size_t idx = (size_t)blockIdx.x * blockDim.x + threadIdx.x;
  if (idx >= (size_t)nedges * (HQ / 8)) return;
  const int k = (int)(idx / (HQ / 8));
  const int hb = (int)(idx % (HQ / 8)) * 8;
  if (mode == 1 && spk[src[k]] != spk[dst[k]]) return;
  const float* xr = x + (size_t)src[k] * HQ + hb;
  float* ar = agg + (size_t)dst[k] * HQ + hb;
#pragma unroll
  for (int j = 0; j < 8; ++j) atomicAdd(&ar[j], xr[j]);
}

__global__ __launch_bounds__(256)
void k_agg_tk(const int* __restrict__ jn, const float* __restrict__ x,
              float* __restrict__ agg)
{
  const size_t idx = (size_t)blockIdx.x * blockDim.x + threadIdx.x;
  if (idx >= (size_t)NQ * (HQ / 8)) return;
  const int n = (int)(idx / (HQ / 8));
  const int hb = (int)(idx % (HQ / 8)) * 8;
  const int b = n / TQ;
  const int jg = b * TQ + jn[n];
  const float* xn = x + (size_t)n * HQ + hb;
  const float* xj = x + (size_t)jg * HQ + hb;
  float* an = agg + (size_t)n * HQ + hb;
  float* aj = agg + (size_t)jg * HQ + hb;
#pragma unroll
  for (int j = 0; j < 8; ++j) {
    atomicAdd(&aj[j], xn[j]);   // edge (n -> jg)
    atomicAdd(&an[j], xj[j]);   // edge (jg -> n)
  }
}

__global__ __launch_bounds__(256)
void k_highpass_pre(const float* __restrict__ x, const float* __restrict__ agg,
                    const float* __restrict__ deg, float* __restrict__ y)
{
  const size_t idx = (size_t)blockIdx.x * blockDim.x + threadIdx.x;
  if (idx >= (size_t)NQ * HQ) return;
  const int n = (int)(idx / HQ);
  y[idx] = x[idx] - agg[idx] / fmaxf(deg[n], 1.f);
}

__global__ __launch_bounds__(256)
void k_uf(const float* __restrict__ hs, const float* __restrict__ lw,
          const float* __restrict__ beta, float* __restrict__ uf)
{
  const size_t idx = (size_t)blockIdx.x * blockDim.x + threadIdx.x;
  const size_t NH = (size_t)NQ * HQ;
  if (idx >= NH) return;
  float w0 = lw[0], w1 = lw[1], w2 = lw[2], w3 = lw[3];
  float mx = fmaxf(fmaxf(w0, w1), fmaxf(w2, w3));
  float e0 = expf(w0 - mx), e1 = expf(w1 - mx), e2 = expf(w2 - mx), e3 = expf(w3 - mx);
  const float zs = e0 + e1 + e2 + e3;
  e0 /= zs; e1 /= zs; e2 /= zs; e3 /= zs;
  const float bt = 1.f / (1.f + expf(-beta[0]));
  const float aggv = e0 * hs[idx] + e1 * hs[NH + idx] + e2 * hs[2 * NH + idx]
                   + e3 * hs[3 * NH + idx];
  uf[idx] = bt * hs[3 * NH + idx] + (1.f - bt) * aggv;
}

// final assembly: out[n, m*1536 + j] with j -> {feats, hyper, uf}
__global__ __launch_bounds__(256)
void k_output(const float* __restrict__ feats, const float* __restrict__ hx,
              const float* __restrict__ uf, float* __restrict__ out)
{
  const size_t total = (size_t)NQ * 3 * (NDQ + 2 * HQ);
  const size_t idx = (size_t)blockIdx.x * blockDim.x + threadIdx.x;
  if (idx >= total) return;
  const int df = NDQ + 2 * HQ;            // 1536
  const int n = (int)(idx / (3 * df));
  const int c = (int)(idx % (3 * df));
  const int m = c / df, j = c % df;
  const int b = n / TQ, t = n % TQ;
  const size_t n3 = (size_t)b * 3 * TQ + (size_t)m * TQ + t;
  float v;
  if (j < NDQ)            v = feats[n3 * NDQ + j];
  else if (j < NDQ + HQ)  v = hx[n3 * HQ + (j - NDQ)];
  else                    v = uf[(size_t)n * HQ + (j - NDQ - HQ)];
  out[idx] = v;
}

// ---------------------------------------------------------------------------
// Host launcher
// ---------------------------------------------------------------------------
extern "C" void kernel_launch(void* const* d_in, const int* in_sizes, int n_in,
                              void* d_out, int out_size, void* d_ws, size_t ws_size,
                              hipStream_t stream) {
  (void)n_in; (void)out_size; (void)ws_size;
  const float* a_in   = (const float*)d_in[0];
  const float* v_in   = (const float*)d_in[1];
  const float* l_in   = (const float*)d_in[2];
  const float* qmask  = (const float*)d_in[3];
  const float* uttW   = (const float*)d_in[4];
  const float* uttB   = (const float*)d_in[5];
  const float* spkE   = (const float*)d_in[6];
  const float* fc1W   = (const float*)d_in[7];
  const float* fc1B   = (const float*)d_in[8];
  const float* hTheta = (const float*)d_in[9];
  const float* hAtt   = (const float*)d_in[10];
  const float* hBias  = (const float*)d_in[11];
  const float* convW  = (const float*)d_in[12];
  const float* convB  = (const float*)d_in[13];
  const float* lw     = (const float*)d_in[14];
  const float* beta   = (const float*)d_in[15];
  const float* heW    = (const float*)d_in[16];
  const float* ewW    = (const float*)d_in[17];
  const float* attr1  = (const float*)d_in[18];
  const float* attr2  = (const float*)d_in[19];
  const float* type1  = (const float*)d_in[20];
  const int* win_src  = (const int*)d_in[21];
  const int* win_dst  = (const int*)d_in[22];
  const int* spk_src  = (const int*)d_in[23];
  const int* spk_dst  = (const int*)d_in[24];
  const int* hyp_node = (const int*)d_in[25];
  const int* hyp_edge = (const int*)d_in[26];
  const int NWIN  = in_sizes[21];
  const int NSPKE = in_sizes[23];
  const int KINC  = in_sizes[25];

  // workspace carve-up
  char* wsp = (char*)d_ws;
  auto alloc = [&](size_t bytes) -> void* {
    void* p = (void*)wsp;
    wsp += (bytes + 15) & ~(size_t)15;
    return p;
  };
  int*   spk   = (int*)  alloc((size_t)NQ * 4);
  float* l2    = (float*)alloc((size_t)NQ * NDQ * 4);
  float* feats = (float*)alloc((size_t)N3Q * NDQ * 4);
  float* ucat  = (float*)alloc((size_t)NQ * 3 * NDQ * 4);
  float* xBuf  = (float*)alloc((size_t)N3Q * HQ * 4);
  float* xpB   = (float*)alloc((size_t)N3Q * HQ * 4);
  float* epB   = (float*)alloc((size_t)MQ * HQ * 4);
  float* eattr = (float*)alloc((size_t)MQ * HQ * 4);
  float* efB   = (float*)alloc((size_t)MQ * HQ * 4);
  float* accB  = (float*)alloc((size_t)N3Q * HQ * 4);
  float* snB   = (float*)alloc((size_t)N3Q * 4);
  float* seB   = (float*)alloc((size_t)MQ * 4);
  float* eexp  = (float*)alloc((size_t)KINC * 4);
  float* zB    = (float*)alloc((size_t)N3Q * 4);
  float* BdegB = (float*)alloc((size_t)MQ * 4);
  float* DdegB = (float*)alloc((size_t)N3Q * 4);
  float* uB    = (float*)alloc((size_t)NQ * HQ * 4);
  float* ninv  = (float*)alloc((size_t)NQ * 4);
  int*   jn    = (int*)  alloc((size_t)NQ * 4);
  float* degB  = (float*)alloc((size_t)NQ * 4);
  float* aggB  = (float*)alloc((size_t)NQ * HQ * 4);
  float* yB    = (float*)alloc((size_t)NQ * HQ * 4);
  float* hsB   = (float*)alloc((size_t)KCONV * NQ * HQ * 4);
  float* ufB   = (float*)alloc((size_t)NQ * HQ * 4);

  // ---- stage 0: speaker + feats/ucat + e_attr ----
  k_spk_l2<<<NQ, 128, 0, stream>>>(l_in, qmask, spkE, l2, spk);
  k_feats<<<CDIV((size_t)NQ * NDQ, 256), 256, 0, stream>>>(l2, a_in, v_in, feats, ucat);
  k_eattr<<<CDIV((size_t)MQ * HQ, 256), 256, 0, stream>>>(attr1, attr2, type1, eattr);

  // ---- x1 = feats @ fc1_w + fc1_b ----
  {
    dim3 g(HQ / 64, N3Q / 32);
    k_gemm_wmma<true, false><<<g, 32, 0, stream>>>(feats, fc1W, fc1B, nullptr,
                                                   xBuf, N3Q, NDQ, HQ);
  }

  // ---- hyper degrees (layer-invariant) ----
  hipMemsetAsync(BdegB, 0, (size_t)MQ * 4, stream);
  hipMemsetAsync(DdegB, 0, (size_t)N3Q * 4, stream);
  k_degrees<<<CDIV(KINC, 256), 256, 0, stream>>>(hyp_node, hyp_edge, heW, BdegB, DdegB, KINC);

  // ---- 3 hypergraph conv layers (in-place on xBuf) ----
  for (int i = 0; i < LHYP; ++i) {
    const float* theta = hTheta + (size_t)i * HQ * HQ;
    const float* atti  = hAtt + (size_t)i * 2 * HQ;
    const float* biasi = hBias + (size_t)i * HQ;
    {
      dim3 g(HQ / 64, N3Q / 32);
      k_gemm_wmma<false, false><<<g, 32, 0, stream>>>(xBuf, theta, nullptr, nullptr,
                                                      xpB, N3Q, HQ, HQ);
    }
    {
      dim3 g(HQ / 64, MQ / 32);
      k_gemm_wmma<false, false><<<g, 32, 0, stream>>>(eattr, theta, nullptr, nullptr,
                                                      epB, MQ, HQ, HQ);
    }
    k_rowdot<<<N3Q, 32, 0, stream>>>(xpB, atti, snB, HQ);
    k_rowdot<<<MQ, 32, 0, stream>>>(epB, atti + HQ, seB, HQ);
    hipMemsetAsync(zB, 0, (size_t)N3Q * 4, stream);
    k_incexp<<<CDIV(KINC, 256), 256, 0, stream>>>(hyp_node, hyp_edge, snB, seB, eexp, zB, KINC);
    hipMemsetAsync(efB, 0, (size_t)MQ * HQ * 4, stream);
    k_scatter_ef<<<CDIV((size_t)KINC * (HQ / 8), 256), 256, 0, stream>>>(
        hyp_node, hyp_edge, eexp, zB, ewW, xpB, efB, KINC);
    k_ef_scale<<<CDIV((size_t)MQ * HQ, 256), 256, 0, stream>>>(heW, BdegB, efB);
    hipMemsetAsync(accB, 0, (size_t)N3Q * HQ * 4, stream);
    k_scatter_out<<<CDIV((size_t)KINC * (HQ / 8), 256), 256, 0, stream>>>(
        hyp_node, hyp_edge, eexp, zB, efB, accB, KINC);
    k_hyper_final<<<CDIV((size_t)N3Q * HQ, 256), 256, 0, stream>>>(accB, DdegB, biasi, xBuf);
  }

  // ---- u = ucat @ utt_fc_w + b ----
  {
    dim3 g(HQ / 64, NQ / 32);
    k_gemm_wmma<true, false><<<g, 32, 0, stream>>>(ucat, uttW, uttB, nullptr,
                                                   uB, NQ, 3 * NDQ, HQ);
  }
  k_norminv<<<NQ, 32, 0, stream>>>(uB, ninv);
  k_simargmax<<<NQ, 64, 0, stream>>>(uB, ninv, jn);

  // ---- high-pass conv degrees (layer-invariant) ----
  hipMemsetAsync(degB, 0, (size_t)NQ * 4, stream);
  k_deg_pairs<<<CDIV(NWIN, 256), 256, 0, stream>>>(win_src, win_dst, spk, 0, degB, NWIN);
  k_deg_pairs<<<CDIV(NSPKE, 256), 256, 0, stream>>>(spk_src, spk_dst, spk, 1, degB, NSPKE);
  k_deg_tk<<<CDIV(NQ, 256), 256, 0, stream>>>(jn, degB);

  // ---- 4 residual high-pass conv layers ----
  const float* xcur = uB;
  for (int k = 0; k < KCONV; ++k) {
    float* hsk = hsB + (size_t)k * NQ * HQ;
    hipMemsetAsync(aggB, 0, (size_t)NQ * HQ * 4, stream);
    k_agg_pairs<<<CDIV((size_t)NWIN * (HQ / 8), 256), 256, 0, stream>>>(
        win_src, win_dst, spk, 0, xcur, aggB, NWIN);
    k_agg_pairs<<<CDIV((size_t)NSPKE * (HQ / 8), 256), 256, 0, stream>>>(
        spk_src, spk_dst, spk, 1, xcur, aggB, NSPKE);
    k_agg_tk<<<CDIV((size_t)NQ * (HQ / 8), 256), 256, 0, stream>>>(jn, xcur, aggB);
    k_highpass_pre<<<CDIV((size_t)NQ * HQ, 256), 256, 0, stream>>>(xcur, aggB, degB, yB);
    {
      dim3 g(HQ / 64, NQ / 32);
      // hs_k = xcur + y @ conv_w[k] + conv_b[k]
      k_gemm_wmma<true, true><<<g, 32, 0, stream>>>(yB, convW + (size_t)k * HQ * HQ,
                                                    convB + (size_t)k * HQ, xcur,
                                                    hsk, NQ, HQ, HQ);
    }
    xcur = hsk;
  }

  // ---- layer mix + final assembly ----
  k_uf<<<CDIV((size_t)NQ * HQ, 256), 256, 0, stream>>>(hsB, lw, beta, ufB);
  k_output<<<CDIV((size_t)NQ * 3 * (NDQ + 2 * HQ), 256), 256, 0, stream>>>(
      feats, xBuf, ufB, (float*)d_out);
}
